// LongContextAttention_42228118455016
// MI455X (gfx1250) — matrix operand order repaired and checked by reading, not compiled, not run on hardware
//
#include <hip/hip_runtime.h>
#include <hip/hip_fp16.h>

typedef __attribute__((ext_vector_type(16))) _Float16 v16h;
typedef __attribute__((ext_vector_type(8)))  float    v8f;

#define HEADS     16
#define HDIM      128
#define SEQ       2048
#define ROWSTRIDE (HEADS * HDIM)   // 2048 floats per sequence position
#define BM        64               // query rows per workgroup (16 per wave)
#define BN        32               // keys per inner iteration (WMMA K=32 for PV)
#define QK_SCALE  0.08838834764831845f  // 1/sqrt(128)

__global__ __launch_bounds__(128)
void flash_attn_fwd_f16wmma(const float* __restrict__ Q,
                            const float* __restrict__ K,
                            const float* __restrict__ V,
                            float* __restrict__ O)
{
    // LDS staging. 36 KB total (WGP has 320 KB).
    __shared__ __align__(32) _Float16 sQ [BM * HDIM];     // 16 KB, f16, scaled
    __shared__ __align__(32) _Float16 sKt[HDIM * BN];     //  8 KB, row-major [t][d]
    __shared__ __align__(32) _Float16 sV [BN * HDIM];     //  8 KB, transposed [d][t]
    __shared__ __align__(32) _Float16 sP [4][16 * BN];    //  4 KB, per-wave P relayout

    const int tid  = threadIdx.x;
    const int w    = tid >> 5;        // wave id 0..3
    const int lane = tid & 31;
    const int lh   = lane >> 4;       // lane half (A/B fragment K-half, C row-half)
    const int ll   = lane & 15;
    const int head = blockIdx.y;
    const int q0   = blockIdx.x * BM;

    // ---- Stage Q tile into LDS as f16, scale folded in ----
    for (int i = tid; i < BM * HDIM; i += 128) {
        const int row = i >> 7, d = i & 127;
        sQ[i] = (_Float16)(Q[(size_t)(q0 + row) * ROWSTRIDE + head * HDIM + d] * QK_SCALE);
    }
    __syncthreads();

    // ---- Resident Q A-fragments: 16x32 f16, lane ll = row, half lh = K 0..15/16..31 ----
    v16h aQ[4];
    const _Float16* qb = &sQ[(16 * w + ll) * HDIM + lh * 16];
#pragma unroll
    for (int dc = 0; dc < 4; ++dc)
        aQ[dc] = *(const v16h*)(qb + dc * 32);

    const int qw = q0 + 16 * w;       // this wave's first query row

    float mrow[8], lrow[8];           // online-softmax stats, one per C-VGPR row
    v8f   accO[8];                    // 16x128 f32 output accumulator (8 N-chunks)
#pragma unroll
    for (int r = 0; r < 8; ++r) {
        mrow[r] = -__builtin_inff();
        lrow[r] = 0.0f;
        accO[r] = (v8f){};
    }

    const int ntiles = q0 / BN + 2;   // causal: keys 0 .. q0+63

    for (int kt = 0; kt < ntiles; ++kt) {
        const int t0 = kt * BN;

        __syncthreads();              // previous iteration done reading sKt/sV
        // ---- Stage K and V (transposed) tiles, f32 -> f16 ----
        for (int i = tid; i < BN * HDIM; i += 128) {
            const int t = i >> 7, d = i & 127;
            const size_t g = (size_t)(t0 + t) * ROWSTRIDE + head * HDIM + d;
            sKt[i]          = (_Float16)K[g];
            sV[d * BN + t]  = (_Float16)V[g];
        }
        // Prefetch next tile into cache hierarchy (global_prefetch_b8).
        if (kt + 1 < ntiles) {
            const int    t  = t0 + BN + (tid >> 2);
            const size_t g  = (size_t)t * ROWSTRIDE + head * HDIM + (tid & 3) * 32;
            __builtin_prefetch(&K[g], 0, 1);
            __builtin_prefetch(&V[g], 0, 1);
        }
        __syncthreads();

        // Waves whose rows lie entirely left of this key tile skip it (causal).
        if (t0 <= qw + 15) {
            // ---- S = Q K^T : two 16x16 C-frags (key halves), 4 d-chunks each ----
            v8f s0 = (v8f){}, s1 = (v8f){};
#pragma unroll
            for (int dc = 0; dc < 4; ++dc) {
                const _Float16* kb = &sKt[ll * HDIM + dc * 32 + lh * 16];   // lane%16 = key, half = K 0..15/16..31
                const v16h b0 = *(const v16h*)(kb);                 // keys t0+0..15
                const v16h b1 = *(const v16h*)(kb + 16 * HDIM);     // keys t0+16..31
                s0 = __builtin_amdgcn_wmma_f32_16x16x32_f16(false, aQ[dc], false, b0,
                                                            (short)0, s0, false, false);
                s1 = __builtin_amdgcn_wmma_f32_16x16x32_f16(false, aQ[dc], false, b1,
                                                            (short)0, s1, false, false);
            }

            // ---- Causal mask (only tiles straddling the diagonal) ----
            if (t0 + BN - 1 > qw) {
#pragma unroll
                for (int r = 0; r < 8; ++r) {
                    const int row = qw + r + 8 * lh;   // C layout: M = r + 8*(lane>=16)
                    const int c0  = t0 + ll;           // N = lane%16
                    if (c0 > row)      s0[r] = -__builtin_inff();
                    if (c0 + 16 > row) s1[r] = -__builtin_inff();
                }
            }

            // ---- Online softmax over the 16-lane row groups ----
#pragma unroll
            for (int r = 0; r < 8; ++r) {
                float mx = fmaxf(s0[r], s1[r]);
#pragma unroll
                for (int sh = 1; sh < 16; sh <<= 1)
                    mx = fmaxf(mx, __shfl_xor(mx, sh, 16));
                const float mnew  = fmaxf(mrow[r], mx);
                const float alpha = __expf(mrow[r] - mnew);
                const float p0 = __expf(s0[r] - mnew);
                const float p1 = __expf(s1[r] - mnew);
                s0[r] = p0; s1[r] = p1;
                float sum = p0 + p1;
#pragma unroll
                for (int sh = 1; sh < 16; sh <<= 1)
                    sum += __shfl_xor(sum, sh, 16);
                lrow[r] = lrow[r] * alpha + sum;
                mrow[r] = mnew;
#pragma unroll
                for (int j = 0; j < 8; ++j) accO[j][r] *= alpha;
            }

            // ---- Relayout P (C-frag) -> A-frag via per-wave LDS (in-order DS) ----
            _Float16* pw = &sP[w][0];
#pragma unroll
            for (int r = 0; r < 8; ++r) {
                const int row = r + 8 * lh;
                pw[row * BN + ll]      = (_Float16)s0[r];
                pw[row * BN + 16 + ll] = (_Float16)s1[r];
            }
            const v16h aP = *(const v16h*)&pw[ll * BN + lh * 16];

            // ---- O += P V : 8 N-chunks of 16, K = 32 keys ----
#pragma unroll
            for (int j = 0; j < 8; ++j) {
                const v16h bV = *(const v16h*)&sV[(j * 16 + ll) * BN + lh * 16];
                accO[j] = __builtin_amdgcn_wmma_f32_16x16x32_f16(false, aP, false, bV,
                                                                 (short)0, accO[j], false, false);
            }
        }
    }

    // ---- Normalize and store ----
#pragma unroll
    for (int r = 0; r < 8; ++r) {
        const int   row = qw + r + 8 * lh;
        const float inv = 1.0f / lrow[r];
#pragma unroll
        for (int j = 0; j < 8; ++j) {
            const int d = j * 16 + ll;
            O[(size_t)row * ROWSTRIDE + head * HDIM + d] = accO[j][r] * inv;
        }
    }
}

extern "C" void kernel_launch(void* const* d_in, const int* /*in_sizes*/, int /*n_in*/,
                              void* d_out, int /*out_size*/, void* /*d_ws*/, size_t /*ws_size*/,
                              hipStream_t stream)
{
    const float* Q = (const float*)d_in[0];
    const float* K = (const float*)d_in[1];
    const float* V = (const float*)d_in[2];
    float*       O = (float*)d_out;

    dim3 grid(SEQ / BM, HEADS, 1);
    dim3 block(128, 1, 1);
    flash_attn_fwd_f16wmma<<<grid, block, 0, stream>>>(Q, K, V, O);
}
